// BoeNet_59038620451478
// MI455X (gfx1250) — compile-verified, matrix operand order checked by reading, not run on hardware
//
#include <hip/hip_runtime.h>
#include <hip/hip_bf16.h>
#include <stdint.h>
#include <stddef.h>

// ---------------- problem constants ----------------
#define NTOK 4096          // B*S = 8*512
#define HD   512           // hidden H
#define ED   512           // embed E
#define VV   32000         // vocab
#define KT16 16            // K/32 for K=512
#define SIB_SCALE 0.04419417382415922f   // 1/sqrt(512)
#define DEP_SCALE 0.01f

// ---------------- WMMA fragment types ----------------
typedef __bf16 v16bf __attribute__((ext_vector_type(16)));
typedef __bf16 v8bf  __attribute__((ext_vector_type(8)));
typedef float  v8f   __attribute__((ext_vector_type(8)));
typedef unsigned int u32x4 __attribute__((ext_vector_type(4)));
typedef int i32x8 __attribute__((ext_vector_type(8)));
typedef int i32x4 __attribute__((ext_vector_type(4)));

__device__ __forceinline__ v8f wmma_bf16(v16bf a, v16bf b, v8f c) {
    return __builtin_amdgcn_wmma_f32_16x16x32_bf16(false, a, false, b,
                                                   (short)0, c, false, false);
}

// ---- fragment-major load: frag = one 16x32 (or 32x16) tile stored as
// ---- 32 lanes x 16 bf16 contiguous (1KB). One 32B read per lane.
__device__ __forceinline__ v16bf load_fragm(const __bf16* F, int KT, int t_outer,
                                            int kt, int lane) {
    const __bf16* p = F + (((size_t)t_outer * KT + kt) << 9) + (lane << 4);
    return *(const v16bf*)p;   // 32B aligned -> 2x global_load_b128
}

// ---- row-major A fragment (16x32 bf16), two 16B vector loads per lane.
__device__ __forceinline__ v16bf load_a_frag(const __bf16* A, int lda,
                                             int row0, int kb, int lane) {
    const int r  = row0 + (lane & 15);
    const int k0 = kb + ((lane >> 4) << 3);
    const __bf16* p = A + (size_t)r * lda + k0;
    const v8bf lo = *(const v8bf*)(p);
    const v8bf hi = *(const v8bf*)(p + 16);
    return __builtin_shufflevector(lo, hi, 0,1,2,3,4,5,6,7,8,9,10,11,12,13,14,15);
}

// ---- gathered A fragment from f32 embedding, vector f32 loads + cvt.
__device__ __forceinline__ v16bf load_a_gather(const float* emb, const int* tokens,
                                               int row0, int kb, int lane) {
    const int r  = row0 + (lane & 15);
    const int k0 = kb + ((lane >> 4) << 3);
    const float* p = emb + (size_t)tokens[r] * ED + k0;
    typedef float v4f __attribute__((ext_vector_type(4)));
    const v4f a0 = *(const v4f*)(p);
    const v4f a1 = *(const v4f*)(p + 4);
    const v4f a2 = *(const v4f*)(p + 16);
    const v4f a3 = *(const v4f*)(p + 20);
    v16bf a;
#pragma unroll
    for (int j = 0; j < 4; ++j) {
        a[j]      = (__bf16)a0[j];
        a[4 + j]  = (__bf16)a1[j];
        a[8 + j]  = (__bf16)a2[j];
        a[12 + j] = (__bf16)a3[j];
    }
    return a;
}

// ---------------- swizzle kernel: f32 row-major KxN -> bf16 B-fragment-major --
__global__ void swizzleB_kernel(const float* __restrict__ W, __bf16* __restrict__ Bf,
                                int K, int N) {
    const int idx  = blockIdx.x * blockDim.x + threadIdx.x;
    const int lane = idx & 31;
    const int frag = idx >> 5;
    const int KT = K >> 5;
    if (frag >= KT * (N >> 4)) return;
    const int kt = frag % KT;
    const int nt = frag / KT;
    const int n  = (nt << 4) + (lane & 15);
    const int k0 = (kt << 5) + ((lane >> 4) << 3);
    __bf16* out = Bf + ((size_t)frag << 9) + (lane << 4);
#pragma unroll
    for (int j = 0; j < 8; ++j) {
        out[j]     = (__bf16)W[(size_t)(k0 + j) * N + n];
        out[8 + j] = (__bf16)W[(size_t)(k0 + 16 + j) * N + n];
    }
}

// ---------------- kernel 1: h0 = emb[tokens] @ Wp + bp ----------------
__global__ __launch_bounds__(256) void h0_kernel(const int* __restrict__ tokens,
                                                 const float* __restrict__ emb,
                                                 const __bf16* __restrict__ WpF,
                                                 const float* __restrict__ bp,
                                                 float* __restrict__ agg,
                                                 __bf16* __restrict__ nodes0) {
    const int wave = (blockIdx.x * blockDim.x + threadIdx.x) >> 5;
    const int lane = threadIdx.x & 31;
    const int mt = wave >> 3;        // 256 row tiles of 16
    const int nt = wave & 7;         // 8 col tiles of 64
    const int row0 = mt * 16, col0 = nt * 64;

    v8f c[4] = {};
    for (int kt = 0; kt < KT16; ++kt) {
        v16bf a = load_a_gather(emb, tokens, row0, kt * 32, lane);
#pragma unroll
        for (int f = 0; f < 4; ++f) {
            v16bf b = load_fragm(WpF, KT16, (col0 >> 4) + f, kt, lane);
            c[f] = wmma_bf16(a, b, c[f]);
        }
    }
    const int laneHi = lane >> 4, nl = lane & 15;
#pragma unroll
    for (int f = 0; f < 4; ++f) {
        const int n = col0 + f * 16 + nl;
        const float bn = bp[n];
#pragma unroll
        for (int r = 0; r < 8; ++r) {
            const int m = r + 8 * laneHi;
            const float v = c[f][r] + bn;
            const size_t o = (size_t)(row0 + m) * HD + n;
            agg[o] = v;
            nodes0[o] = (__bf16)v;
        }
    }
}

// ---------------- gate kernel ----------------
__global__ __launch_bounds__(256) void gate_kernel(const __bf16* __restrict__ nodes,
                                                   const float* __restrict__ dep_d,
                                                   const float* __restrict__ Wg,
                                                   const float* __restrict__ bg,
                                                   const float* __restrict__ ePrev,
                                                   float* __restrict__ eOut,
                                                   int isFirst) {
    const int row  = (blockIdx.x * blockDim.x + threadIdx.x) >> 5;
    const int lane = threadIdx.x & 31;
    const __bf16* nr = nodes + (size_t)row * HD;
    float g = 0.f;
    for (int e = lane; e < HD; e += 32)
        g += ((float)nr[e] + DEP_SCALE * dep_d[e]) * Wg[e];
#pragma unroll
    for (int off = 16; off; off >>= 1) g += __shfl_xor(g, off, 32);
    if (lane == 0) {
        g += bg[0];
        const float act = isFirst ? 1.f : ePrev[row >> 1];
        eOut[row] = (g > 0.f && act > 0.5f) ? 1.f : 0.f;
    }
}

// ---------------- tree kernel ----------------
__global__ __launch_bounds__(256) void tree_kernel(const __bf16* __restrict__ nodesIn,
                                                   const __bf16* __restrict__ WcF,
                                                   const float* __restrict__ bc,
                                                   const float* __restrict__ sib,
                                                   const float* __restrict__ eMask,
                                                   float* __restrict__ agg,
                                                   __bf16* __restrict__ nodesOut,
                                                   int logn) {
    const int wave = (blockIdx.x * blockDim.x + threadIdx.x) >> 5;
    const int lane = threadIdx.x & 31;
    const int rt = wave >> 4;
    const int ct = wave & 15;
    const int row0 = rt * 16, col0 = ct * 32;
    const int ntBase = ct * 2;

    v8f c00 = {}, c01 = {}, c10 = {}, c11 = {};
    for (int kt = 0; kt < KT16; ++kt) {
        v16bf a = load_a_frag(nodesIn, HD, row0, kt * 32, lane);
        v16bf b;
        b = load_fragm(WcF, KT16, ntBase,          kt, lane); c00 = wmma_bf16(a, b, c00);
        b = load_fragm(WcF, KT16, ntBase + 1,      kt, lane); c01 = wmma_bf16(a, b, c01);
        b = load_fragm(WcF, KT16, 32 + ntBase,     kt, lane); c10 = wmma_bf16(a, b, c10);
        b = load_fragm(WcF, KT16, 32 + ntBase + 1, kt, lane); c11 = wmma_bf16(a, b, c11);
    }
    const int laneHi = lane >> 4, nl = lane & 15;
#pragma unroll
    for (int f = 0; f < 2; ++f) {
        const int col = col0 + f * 16 + nl;
        const float b0 = bc[col]      + SIB_SCALE * sib[col];
        const float b1 = bc[HD + col] + SIB_SCALE * sib[HD + col];
        const v8f cf0 = f ? c01 : c00;
        const v8f cf1 = f ? c11 : c10;
        float acc = 0.f; int curTok = -1;
#pragma unroll
        for (int r = 0; r < 8; ++r) {
            const int grow = row0 + r + 8 * laneHi;   // parent row
            const float ch0 = cf0[r] + b0;
            const float ch1 = cf1[r] + b1;
            if (nodesOut) {
                const size_t o = (size_t)(2 * grow) * HD + col;
                nodesOut[o]      = (__bf16)ch0;
                nodesOut[o + HD] = (__bf16)ch1;
            }
            const int tok = grow >> logn;
            if (tok != curTok) {
                if (curTok >= 0) agg[(size_t)curTok * HD + col] += acc;
                curTok = tok; acc = 0.f;
            }
            acc += eMask[grow] * (ch0 + ch1);
        }
        agg[(size_t)curTok * HD + col] += acc;
    }
}

// ---------------- pooling kernel: writes pooled in A-fragment layout ---------
__global__ void pool_kernel(const float* __restrict__ agg,
                            const float* __restrict__ e0, const float* __restrict__ e1,
                            const float* __restrict__ e2, __bf16* __restrict__ pooledF) {
    const int idx = blockIdx.x * blockDim.x + threadIdx.x;
    if (idx >= NTOK * HD) return;
    const int m = idx >> 9;
    const int k = idx & 511;
    const float cnt = 1.f + 2.f * (e0[m]
                    + e1[2 * m] + e1[2 * m + 1]
                    + e2[4 * m] + e2[4 * m + 1] + e2[4 * m + 2] + e2[4 * m + 3]);
    const float v = agg[idx] / fmaxf(cnt, 1e-8f);
    const int mt = m >> 4, kt = k >> 5, kp = k & 31;
    const int laneHi = (kp >> 3) & 1;
    const int j = (kp & 7) + (((kp >> 4) & 1) << 3);
    const int lane = (m & 15) + (laneHi << 4);
    pooledF[(((size_t)mt * KT16 + kt) << 9) + (lane << 4) + j] = (__bf16)v;
}

// ---------------- logits kernel: pooled @ Wout + bout ----------------
// Block tile: 128 rows x 64 cols (8 waves x 16 rows). The 4 column-tiles'
// fragment panels are CONTIGUOUS 64KB in WoutF -> one 1-D TDM descriptor
// stages them into LDS; all 8 waves then feed WMMA from ds_load_b128.
#define LOGITS_PANEL_ELEMS (4 * KT16 * 512)            // 32768 bf16 = 64KB

__global__ __launch_bounds__(256) void logits_kernel(const __bf16* __restrict__ pooledF,
                                                     const __bf16* __restrict__ WoutF,
                                                     const float* __restrict__ bout,
                                                     float* __restrict__ out) {
    __shared__ __bf16 ldsB[LOGITS_PANEL_ELEMS];        // 64 KB panel
    const int bid  = blockIdx.x;
    const int brow = bid / 500;            // 32 row blocks (128 rows each)
    const int bcol = bid % 500;            // 500 col blocks (64 cols each)
    const int waveId = threadIdx.x >> 5;
    const int lane   = threadIdx.x & 31;
    const int nt0 = bcol * 4;
    const __bf16* gPanel = WoutF + (((size_t)nt0 * KT16) << 9);

#if __has_builtin(__builtin_amdgcn_tensor_load_to_lds)
    if (waveId == 0) {
        // ---- Tensor Data Mover: 1-D copy of 64KB panel into LDS ----
        const unsigned long long gaddr = (unsigned long long)(uintptr_t)gPanel;
        const unsigned int ldsaddr = (unsigned int)(uintptr_t)(&ldsB[0]);
        const unsigned int nUnits = (LOGITS_PANEL_ELEMS * 2) / 8;   // 8192 x 8B
        u32x4 g0;
        g0[0] = 1u;                                        // count=1 (user D#)
        g0[1] = ldsaddr;                                   // lds_addr
        g0[2] = (unsigned int)(gaddr & 0xFFFFFFFFu);       // global_addr lo
        g0[3] = (unsigned int)((gaddr >> 32) & 0x1FFFFFFu) // global_addr hi
              | 0x80000000u;                               // type=2 ("image")
        i32x8 g1;
        g1[0] = (3 << 16);                                 // data_size=8B
        g1[1] = (int)((nUnits & 0xFFFFu) << 16);           // tensor_dim0 lo16
        g1[2] = (int)((nUnits >> 16) & 0xFFFFu) | (1 << 16); // dim0 hi16, dim1=1
        g1[3] = (int)(nUnits << 16);                       // tile_dim0
        g1[4] = 1;                                         // tile_dim1=1, tile_dim2=0
        g1[5] = (int)nUnits;                               // dim0_stride lo32
        g1[6] = 0;
        g1[7] = 0;
        i32x4 g2 = {0, 0, 0, 0}, g3 = {0, 0, 0, 0};
        i32x8 g4 = {0, 0, 0, 0, 0, 0, 0, 0};               // unused groups (2-D tile)
        __builtin_amdgcn_tensor_load_to_lds(g0, g1, g2, g3, g4, 0);
        __builtin_amdgcn_s_wait_tensorcnt((short)0);
    }
#else
    {   // fallback: cooperative b128 global->LDS copy
        typedef __bf16 v8b __attribute__((ext_vector_type(8)));
        for (int i = threadIdx.x; i < LOGITS_PANEL_ELEMS / 8; i += 256)
            *(v8b*)(&ldsB[i * 8]) = *(const v8b*)(gPanel + i * 8);
    }
#endif
    __syncthreads();

    const int mt = brow * 8 + waveId;
    const int row0 = mt * 16;
    const int col0 = bcol * 64;

    v8f c[4] = {};
    for (int kt = 0; kt < KT16; ++kt) {
        v16bf a = load_fragm(pooledF, KT16, mt, kt, lane);
#pragma unroll
        for (int f = 0; f < 4; ++f) {
            const __bf16* p = &ldsB[(((f * KT16) + kt) << 9) + (lane << 4)];
            v16bf b = *(const v16bf*)p;     // 2x ds_load_b128
            c[f] = wmma_bf16(a, b, c[f]);
        }
    }
    const int laneHi = lane >> 4, nl = lane & 15;
#pragma unroll
    for (int f = 0; f < 4; ++f) {
        const int n = col0 + f * 16 + nl;
        const float bn = bout[n];
#pragma unroll
        for (int r = 0; r < 8; ++r) {
            const int m = r + 8 * laneHi;
            out[(size_t)(row0 + m) * VV + n] = c[f][r] + bn;
        }
    }
}

// ---------------- host-side orchestration ----------------
extern "C" void kernel_launch(void* const* d_in, const int* in_sizes, int n_in,
                              void* d_out, int out_size, void* d_ws, size_t ws_size,
                              hipStream_t stream) {
    const int*   tokens = (const int*)  d_in[0];
    const float* emb    = (const float*)d_in[1];
    const float* Wp     = (const float*)d_in[2];
    const float* bp     = (const float*)d_in[3];
    const float* Wc     = (const float*)d_in[4];
    const float* bc     = (const float*)d_in[5];
    const float* Wg     = (const float*)d_in[6];
    const float* bg     = (const float*)d_in[7];
    const float* dep    = (const float*)d_in[8];
    const float* sib    = (const float*)d_in[9];
    const float* Wout   = (const float*)d_in[10];
    const float* bout   = (const float*)d_in[11];
    float* out = (float*)d_out;

    char* w = (char*)d_ws;
    size_t off = 0;
    auto alloc = [&](size_t bytes) -> void* {
        void* p = w + off;
        off = (off + bytes + 255) & ~(size_t)255;
        return p;
    };
    __bf16* WpF    = (__bf16*)alloc((size_t)ED * HD * 2);
    __bf16* WcF    = (__bf16*)alloc((size_t)HD * 2 * HD * 2);
    __bf16* WoutF  = (__bf16*)alloc((size_t)HD * VV * 2);
    float*  agg    = (float*) alloc((size_t)NTOK * HD * 4);
    __bf16* nodes0 = (__bf16*)alloc((size_t)NTOK * HD * 2);
    __bf16* nodes1 = (__bf16*)alloc((size_t)NTOK * 2 * HD * 2);
    __bf16* nodes2 = (__bf16*)alloc((size_t)NTOK * 4 * HD * 2);
    float*  e0     = (float*) alloc((size_t)NTOK * 4);
    float*  e1     = (float*) alloc((size_t)NTOK * 2 * 4);
    float*  e2     = (float*) alloc((size_t)NTOK * 4 * 4);
    __bf16* pooledF= (__bf16*)alloc((size_t)NTOK * HD * 2);

    // 0) weight swizzle to bf16 fragment-major layout
    {
        int thrWp = (ED / 32) * (HD / 16) * 32;
        int thrWc = (HD / 32) * (2 * HD / 16) * 32;
        int thrWo = (HD / 32) * (VV / 16) * 32;
        swizzleB_kernel<<<(thrWp + 255) / 256, 256, 0, stream>>>(Wp, WpF, ED, HD);
        swizzleB_kernel<<<(thrWc + 255) / 256, 256, 0, stream>>>(Wc, WcF, HD, 2 * HD);
        swizzleB_kernel<<<(thrWo + 255) / 256, 256, 0, stream>>>(Wout, WoutF, HD, VV);
    }

    // 1) h0 GEMM (gathered A)
    h0_kernel<<<256, 256, 0, stream>>>(tokens, emb, WpF, bp, agg, nodes0);

    // 2) tree depths (gate -> expand/aggregate)
    gate_kernel<<<4096 / 8, 256, 0, stream>>>(nodes0, dep + 0 * HD, Wg, bg, nullptr, e0, 1);
    tree_kernel<<<4096 / 8, 256, 0, stream>>>(nodes0, WcF, bc, sib, e0, agg, nodes1, 0);
    gate_kernel<<<8192 / 8, 256, 0, stream>>>(nodes1, dep + 1 * HD, Wg, bg, e0, e1, 0);
    tree_kernel<<<8192 / 8, 256, 0, stream>>>(nodes1, WcF, bc, sib, e1, agg, nodes2, 1);
    gate_kernel<<<16384 / 8, 256, 0, stream>>>(nodes2, dep + 2 * HD, Wg, bg, e1, e2, 0);
    tree_kernel<<<16384 / 8, 256, 0, stream>>>(nodes2, WcF, bc, sib, e2, agg, nullptr, 2);

    // 3) pooling (writes A-fragment layout for the logits GEMM)
    pool_kernel<<<(NTOK * HD + 255) / 256, 256, 0, stream>>>(agg, e0, e1, e2, pooledF);

    // 4) logits GEMM: 32 row-blocks x 500 col-blocks, TDM-staged B panels
    logits_kernel<<<16000, 256, 0, stream>>>(pooledF, WoutF, bout, out);
}
